// multi_modal_layer_52879637348668
// MI455X (gfx1250) — compile-verified, matrix operand order checked by reading, not run on hardware
//
#include <hip/hip_runtime.h>
#include <math.h>

// ---------------------------------------------------------------------------
// Types for CDNA5 WMMA (wave32)
// ---------------------------------------------------------------------------
typedef __attribute__((ext_vector_type(16))) __bf16 v16bf;
typedef __attribute__((ext_vector_type(8)))  float  v8f;

union FragBF16 {
    v16bf v;
    unsigned short u[16];
    uint4 q[2];
};

__device__ __forceinline__ unsigned short f32_to_bf16(float f) {
    unsigned int x = __float_as_uint(f);
    unsigned int r = x + 0x7FFFu + ((x >> 16) & 1u);   // round to nearest even
    return (unsigned short)(r >> 16);
}

// Pack two f32 into two bf16 (truncating) with a single v_perm_b32:
// result = { hi[31:16], lo[31:16] }
__device__ __forceinline__ unsigned int pack_bf16x2(float lo, float hi) {
    return __builtin_amdgcn_perm(__float_as_uint(hi), __float_as_uint(lo), 0x07060302u);
}

__device__ __forceinline__ float sigm(float x) { return 1.0f / (1.0f + expf(-x)); }

// ---------------------------------------------------------------------------
// Generic batched GEMM:  C[M,N] = A[M,K] @ op(B) + bias (+ C if accum)
//   B_KN=false : B stored [N,K] row-major (PyTorch Linear weight, C = A @ B^T)
//   B_KN=true  : B stored [K,N] row-major (C = A @ B)
// bf16 WMMA 16x16x32, fp32 accumulate; f32 sources converted while staging LDS.
// Block: 256 threads = 8 waves (2Mx4N); block tile 64x128; wave tile 32x32
// (2 A-frags x 2 B-frags -> 4 WMMAs per K-chunk per wave).
// Double-buffered LDS: one barrier per K-chunk; next chunk's global loads
// overlap current chunk's WMMAs.
// ---------------------------------------------------------------------------
template <bool B_KN>
__global__ __launch_bounds__(256)
void gemm_bf16_wmma(const float* __restrict__ A, const float* __restrict__ B,
                    const float* __restrict__ bias, float* __restrict__ C,
                    int M, int N, int K,
                    long long lda, long long ldb, long long ldc,
                    long long strideA, long long strideB, long long strideC,
                    int accum) {
    __shared__ alignas(16) unsigned short As[2][64][40];    // [buf][m][k], 80B row stride
    __shared__ alignas(16) unsigned short Bs[2][128][40];   // [buf][n][k]

    const int tid  = threadIdx.x;
    const int lane = tid & 31;
    const int wave = tid >> 5;
    const int wm   = wave >> 2;       // 0..1 (M waves)
    const int wn   = wave & 3;        // 0..3 (N waves)
    const int m0   = blockIdx.y * 64;
    const int n0   = blockIdx.x * 128;

    const long long zA = (long long)blockIdx.z * strideA;
    const long long zB = (long long)blockIdx.z * strideB;
    const long long zC = (long long)blockIdx.z * strideC;

    v8f acc00 = {0.f,0.f,0.f,0.f,0.f,0.f,0.f,0.f};
    v8f acc01 = acc00, acc10 = acc00, acc11 = acc00;

    const int lr = lane & 15;         // row/col within 16-wide tile
    const int hs = lane >> 4;         // half select (K sub-group)

    const bool mInt = (m0 + 64 <= M);
    const bool nInt = (n0 + 128 <= N);

    // ---- tile staging (fast path: float4 loads + packed 8B LDS stores) ----
    auto stageA = [&](int buf, int k0) {
        if (mInt && (k0 + 32 <= K)) {
#pragma unroll
            for (int j = 0; j < 2; ++j) {
                int i  = tid + j * 256;               // 64 rows x 8 quads
                int r  = i >> 3, c4 = i & 7;
                const float4 v = *(const float4*)&A[zA + (long long)(m0 + r) * lda + k0 + 4 * c4];
                *(uint2*)&As[buf][r][4 * c4] = make_uint2(pack_bf16x2(v.x, v.y), pack_bf16x2(v.z, v.w));
            }
        } else {
            for (int i = tid; i < 64 * 32; i += 256) {
                int r = i >> 5, kk = i & 31;
                int gm = m0 + r, gk = k0 + kk;
                float v = (gm < M && gk < K) ? A[zA + (long long)gm * lda + gk] : 0.0f;
                As[buf][r][kk] = f32_to_bf16(v);
            }
        }
    };
    auto stageB = [&](int buf, int k0) {
        if (nInt && (k0 + 32 <= K)) {
            if (B_KN) {
                // B[k][n]: float4 along n, scatter 4 ushorts into LDS rows
#pragma unroll
                for (int j = 0; j < 4; ++j) {
                    int i  = tid + j * 256;           // 32 k x 32 n-groups
                    int kk = i >> 5, ng = i & 31;
                    const float4 v = *(const float4*)&B[zB + (long long)(k0 + kk) * ldb + n0 + 4 * ng];
                    Bs[buf][4 * ng + 0][kk] = f32_to_bf16(v.x);
                    Bs[buf][4 * ng + 1][kk] = f32_to_bf16(v.y);
                    Bs[buf][4 * ng + 2][kk] = f32_to_bf16(v.z);
                    Bs[buf][4 * ng + 3][kk] = f32_to_bf16(v.w);
                }
            } else {
#pragma unroll
                for (int j = 0; j < 4; ++j) {
                    int i  = tid + j * 256;           // 128 rows x 8 quads
                    int r  = i >> 3, c4 = i & 7;
                    const float4 v = *(const float4*)&B[zB + (long long)(n0 + r) * ldb + k0 + 4 * c4];
                    *(uint2*)&Bs[buf][r][4 * c4] = make_uint2(pack_bf16x2(v.x, v.y), pack_bf16x2(v.z, v.w));
                }
            }
        } else {
            for (int i = tid; i < 128 * 32; i += 256) {
                int n = i >> 5, kk = i & 31;
                int gn = n0 + n, gk = k0 + kk;
                float v = 0.0f;
                if (gn < N && gk < K)
                    v = B_KN ? B[zB + (long long)gk * ldb + gn]
                             : B[zB + (long long)gn * ldb + gk];
                Bs[buf][n][kk] = f32_to_bf16(v);
            }
        }
    };

    const int nk = (K + 31) / 32;
    stageA(0, 0);
    stageB(0, 0);

    for (int c = 0; c < nk; ++c) {
        const int cur = c & 1;
        __syncthreads();   // staging of `cur` complete (and prior reads of `cur^1` done)

        // ---- fragments from current buffer: two 16B LDS loads each ----
        FragBF16 fa0, fa1, fb0, fb1;
        {
            const uint4* r0 = (const uint4*)As[cur][wm * 32 + lr];
            const uint4* r1 = (const uint4*)As[cur][wm * 32 + 16 + lr];
            fa0.q[0] = r0[hs]; fa0.q[1] = r0[2 + hs];
            fa1.q[0] = r1[hs]; fa1.q[1] = r1[2 + hs];
            const uint4* c0 = (const uint4*)Bs[cur][wn * 32 + lr];
            const uint4* c1 = (const uint4*)Bs[cur][wn * 32 + 16 + lr];
            fb0.q[0] = c0[hs]; fb0.q[1] = c0[2 + hs];
            fb1.q[0] = c1[hs]; fb1.q[1] = c1[2 + hs];
        }

        // ---- stage next chunk into the other buffer (overlaps WMMAs below) ----
        if (c + 1 < nk) {
            stageA(cur ^ 1, (c + 1) * 32);
            stageB(cur ^ 1, (c + 1) * 32);
        }
        // ---- prefetch one chunk further ahead (global_prefetch_b8) ----
        if (c + 2 < nk) {
            int k2 = (c + 2) * 32;
            int pr = m0 + (tid & 63);
            if (pr >= M) pr = M - 1;
            __builtin_prefetch(&A[zA + (long long)pr * lda + k2], 0, 1);
            int pn = n0 + (tid >> 1);
            if (pn >= N) pn = N - 1;
            __builtin_prefetch(B_KN ? &B[zB + (long long)k2 * ldb + pn]
                                    : &B[zB + (long long)pn * ldb + k2], 0, 1);
        }

        acc00 = __builtin_amdgcn_wmma_f32_16x16x32_bf16(false, fa0.v, false, fb0.v, (short)0, acc00, false, false);
        acc01 = __builtin_amdgcn_wmma_f32_16x16x32_bf16(false, fa0.v, false, fb1.v, (short)0, acc01, false, false);
        acc10 = __builtin_amdgcn_wmma_f32_16x16x32_bf16(false, fa1.v, false, fb0.v, (short)0, acc10, false, false);
        acc11 = __builtin_amdgcn_wmma_f32_16x16x32_bf16(false, fa1.v, false, fb1.v, (short)0, acc11, false, false);
    }

    // ---- store 4 sub-tiles; C layout: lanes 0-15 -> M=r, 16-31 -> M=r+8; N=lane&15 ----
    auto storeTile = [&](const v8f& a, int ti, int tj) {
#pragma unroll
        for (int r = 0; r < 8; ++r) {
            int gm = m0 + wm * 32 + ti * 16 + r + hs * 8;
            int gn = n0 + wn * 32 + tj * 16 + lr;
            if (gm < M && gn < N) {
                long long ci = zC + (long long)gm * ldc + gn;
                float v = a[r];
                if (bias)  v += bias[gn];
                if (accum) v += C[ci];
                C[ci] = v;
            }
        }
    };
    storeTile(acc00, 0, 0);
    storeTile(acc01, 0, 1);
    storeTile(acc10, 1, 0);
    storeTile(acc11, 1, 1);
}

// ---------------------------------------------------------------------------
// Pointwise / reduction helper kernels
// ---------------------------------------------------------------------------
__global__ void zero_kernel(float* p, long long n) {
    long long i = (long long)blockIdx.x * blockDim.x + threadIdx.x;
    if (i < n) p[i] = 0.0f;
}

__global__ void bias_comb_kernel(const float* a, const float* b, float* out, int n) {
    int i = blockIdx.x * blockDim.x + threadIdx.x;
    if (i < n) out[i] = a[i] + b[i];
}

// mean_feat[b][s][j]: j<100 -> avgpool(att_tempo), j>=100 -> avgpool(att_motion)
__global__ void mean_pool_kernel(const float* __restrict__ att_tempo,
                                 const float* __restrict__ att_motion,
                                 float* __restrict__ out) {
    int idx = blockIdx.x * blockDim.x + threadIdx.x;  // 64*2*200
    if (idx >= 64 * 2 * 200) return;
    int b = idx / 400, rem = idx % 400, s = rem / 200, j = rem % 200;
    const float* src = (j < 100) ? att_tempo : att_motion;
    int jj = j % 100;
    float sum = 0.0f;
    for (int r = 0; r < 10; ++r)
        for (int c = 0; c < 10; ++c)
            sum += src[((long long)(b * 20 + s * 10 + r)) * 1000 + jj * 10 + c];
    out[idx] = sum * 0.01f;
}

// One block per row (1280). Builds mm_inp[r][0:2300] = [emb | tout*w | mout*w],
// w = softmax(mout row) (reference uses temp_mout's softmax for BOTH).
__global__ __launch_bounds__(256)
void softmax_mul_concat_kernel(const float* __restrict__ emb,
                               const float* __restrict__ temp_mout,
                               const float* __restrict__ temp_tout,
                               float* __restrict__ mm_inp) {
    __shared__ float red[256];
    const int row = blockIdx.x, tid = threadIdx.x;
    const float* mo = temp_mout + (long long)row * 1000;
    const float* to = temp_tout + (long long)row * 1000;
    float* dst = mm_inp + (long long)row * 2300;

    for (int j = tid; j < 300; j += 256) dst[j] = emb[(long long)row * 300 + j];

    float m = -INFINITY;
    for (int j = tid; j < 1000; j += 256) m = fmaxf(m, mo[j]);
    red[tid] = m; __syncthreads();
    for (int o = 128; o > 0; o >>= 1) { if (tid < o) red[tid] = fmaxf(red[tid], red[tid + o]); __syncthreads(); }
    const float mx = red[0]; __syncthreads();

    float s = 0.0f;
    for (int j = tid; j < 1000; j += 256) s += expf(mo[j] - mx);
    red[tid] = s; __syncthreads();
    for (int o = 128; o > 0; o >>= 1) { if (tid < o) red[tid] += red[tid + o]; __syncthreads(); }
    const float inv = 1.0f / red[0]; __syncthreads();

    for (int j = tid; j < 1000; j += 256) {
        float w = expf(mo[j] - mx) * inv;
        dst[300 + j]  = to[j] * w;
        dst[1300 + j] = mo[j] * w;
    }
}

// LSTM cell pointwise: gates [64,4096] (i,f,g,o), c/h [64,1024]
__global__ void lstm_cell_kernel(const float* __restrict__ gates,
                                 float* __restrict__ c, float* __restrict__ h,
                                 float* __restrict__ act, long long act_stride) {
    int idx = blockIdx.x * blockDim.x + threadIdx.x;  // 64*1024
    if (idx >= 64 * 1024) return;
    int b = idx >> 10, n = idx & 1023;
    const float* g = gates + (long long)b * 4096;
    float gi = g[n], gf = g[1024 + n], gg = g[2048 + n], go = g[3072 + n];
    float cn = sigm(gf) * c[idx] + sigm(gi) * tanhf(gg);
    float hn = sigm(go) * tanhf(cn);
    c[idx] = cn;
    h[idx] = hn;
    if (act) act[(long long)b * act_stride + n] = tanhf(hn);
}

// out[row] = dot(src[row, 0:cols], vec)   (one block per row)
__global__ __launch_bounds__(256)
void rowdot_kernel(const float* __restrict__ src, const float* __restrict__ vec,
                   float* __restrict__ out, int cols) {
    __shared__ float red[256];
    const int row = blockIdx.x, tid = threadIdx.x;
    const float* p = src + (long long)row * cols;
    float s = 0.0f;
    for (int j = tid; j < cols; j += 256) s += p[j] * vec[j];
    red[tid] = s; __syncthreads();
    for (int o = 128; o > 0; o >>= 1) { if (tid < o) red[tid] += red[tid + o]; __syncthreads(); }
    if (tid == 0) out[row] = red[0];
}

// Masked softmax attention weights. One block per (b,s) of 64*20.
// wmat[b,s,f] = softmax_f( eproj[b,s] + fproj[b,f] + bias ) over f < lens[b], else 0.
__global__ __launch_bounds__(256)
void attn_weights_kernel(const float* __restrict__ eproj,
                         const float* __restrict__ fproj,
                         const int* __restrict__ lens,
                         const float* __restrict__ bias_scalar,
                         float* __restrict__ wmat) {
    __shared__ float red[256];
    const int bs = blockIdx.x, tid = threadIdx.x;
    const int b = bs / 20;
    const int len = lens[b];
    const float e = eproj[bs];
    const float bb = *bias_scalar;

    bool valid = (tid < 200) && (tid < len);
    float score = valid ? (e + fproj[(long long)b * 200 + tid] + bb) : -INFINITY;

    red[tid] = score; __syncthreads();
    for (int o = 128; o > 0; o >>= 1) { if (tid < o) red[tid] = fmaxf(red[tid], red[tid + o]); __syncthreads(); }
    const float mx = red[0]; __syncthreads();

    float ex = valid ? expf(score - mx) : 0.0f;
    red[tid] = ex; __syncthreads();
    for (int o = 128; o > 0; o >>= 1) { if (tid < o) red[tid] += red[tid + o]; __syncthreads(); }
    const float inv = 1.0f / red[0]; __syncthreads();

    if (tid < 200) wmat[(long long)bs * 200 + tid] = ex * inv;
}

// Afin[1280, 3024] = [ act(1024) | att_t2(1000) | att_m2(1000) ]
__global__ void concat_final_kernel(const float* __restrict__ act,
                                    const float* __restrict__ att_t2,
                                    const float* __restrict__ att_m2,
                                    float* __restrict__ Afin) {
    long long idx = (long long)blockIdx.x * blockDim.x + threadIdx.x;
    if (idx >= 1280LL * 3024LL) return;
    int r = (int)(idx / 3024), cidx = (int)(idx % 3024);
    float v;
    if (cidx < 1024)      v = act[(long long)r * 1024 + cidx];
    else if (cidx < 2024) v = att_t2[(long long)r * 1000 + (cidx - 1024)];
    else                  v = att_m2[(long long)r * 1000 + (cidx - 2024)];
    Afin[idx] = v;
}

// In-place log_softmax over rows of length `cols`. One block per row.
__global__ __launch_bounds__(256)
void log_softmax_kernel(float* __restrict__ x, int cols) {
    __shared__ float red[256];
    const int row = blockIdx.x, tid = threadIdx.x;
    float* p = x + (long long)row * cols;

    float m = -INFINITY;
    for (int j = tid; j < cols; j += 256) m = fmaxf(m, p[j]);
    red[tid] = m; __syncthreads();
    for (int o = 128; o > 0; o >>= 1) { if (tid < o) red[tid] = fmaxf(red[tid], red[tid + o]); __syncthreads(); }
    const float mx = red[0]; __syncthreads();

    float s = 0.0f;
    for (int j = tid; j < cols; j += 256) s += expf(p[j] - mx);
    red[tid] = s; __syncthreads();
    for (int o = 128; o > 0; o >>= 1) { if (tid < o) red[tid] += red[tid + o]; __syncthreads(); }
    const float lse = mx + logf(red[0]); __syncthreads();

    for (int j = tid; j < cols; j += 256) p[j] = p[j] - lse;
}

// ---------------------------------------------------------------------------
// Host-side orchestration
// ---------------------------------------------------------------------------
static inline dim3 gemm_grid(int M, int N, int batch) {
    return dim3((unsigned)((N + 127) / 128), (unsigned)((M + 63) / 64), (unsigned)batch);
}

extern "C" void kernel_launch(void* const* d_in, const int* in_sizes, int n_in,
                              void* d_out, int out_size, void* d_ws, size_t ws_size,
                              hipStream_t stream) {
    (void)in_sizes; (void)n_in; (void)out_size; (void)ws_size;

    // Inputs (setup_inputs dict order)
    const float* mfeat      = (const float*)d_in[0];   // [64,200,1000]
    const float* tfeat      = (const float*)d_in[1];   // [64,200,1000]
    const float* att_motion = (const float*)d_in[2];   // [64,20,1000]
    const float* att_tempo  = (const float*)d_in[3];   // [64,20,1000]
    const float* embeddings = (const float*)d_in[4];   // [64,20,300]
    const int*   mlens      = (const int*)d_in[5];     // [64]
    const int*   tlens      = (const int*)d_in[6];     // [64]
    /* seq_batch_lens d_in[7] unused */
    const float* W_im  = (const float*)d_in[8];        // [1000,1000]
    const float* b_im  = (const float*)d_in[9];
    const float* W_it  = (const float*)d_in[10];
    const float* b_it  = (const float*)d_in[11];
    const float* W_mm  = (const float*)d_in[12];       // [1024,2300]
    const float* b_mm  = (const float*)d_in[13];
    const float* W_li  = (const float*)d_in[14];       // [1024,200]
    const float* b_li  = (const float*)d_in[15];
    const float* Wih   = (const float*)d_in[16];       // [4096,1024]
    const float* Whh   = (const float*)d_in[17];       // [4096,1024]
    const float* bih   = (const float*)d_in[18];
    const float* bhh   = (const float*)d_in[19];
    const float* w_ms_e = (const float*)d_in[20];      // [1024]
    const float* w_ms_f = (const float*)d_in[21];      // [1000]
    const float* b_ms   = (const float*)d_in[22];      // scalar
    const float* w_ts_e = (const float*)d_in[23];
    const float* w_ts_f = (const float*)d_in[24];
    const float* b_ts   = (const float*)d_in[25];      // scalar
    const float* W_out  = (const float*)d_in[26];      // [20000,3024]
    const float* b_out  = (const float*)d_in[27];

    float* out = (float*)d_out;                        // [1280,20000]

    // Workspace bump allocator (floats)
    float* ws = (float*)d_ws;
    size_t off = 0;
    auto alloc = [&](size_t n) { float* p = ws + off; off += n; return p; };

    float* temp_mout = alloc(1280 * 1000);
    float* temp_tout = alloc(1280 * 1000);
    float* mm_inp    = alloc((size_t)1280 * 2300);
    float* mm_out    = alloc(1280 * 1024);
    float* mean_feat = alloc(128 * 200);
    float* m0        = alloc(128 * 1024);
    float* gates     = alloc(64 * 4096);
    float* hbuf      = alloc(64 * 1024);
    float* cbuf      = alloc(64 * 1024);
    float* act       = alloc(1280 * 1024);
    float* eproj     = alloc(1280);
    float* fproj     = alloc(64 * 200);
    float* wmat      = alloc((size_t)64 * 20 * 200);
    float* att_m2    = alloc(1280 * 1000);
    float* att_t2    = alloc(1280 * 1000);
    float* bcomb     = alloc(4096);
    float* Afin      = alloc((size_t)1280 * 3024);

    // ---- Stage 1: modality linears (WMMA GEMM, C = A @ W^T + b) ----
    gemm_bf16_wmma<false><<<gemm_grid(1280, 1000, 1), 256, 0, stream>>>(
        att_motion, W_im, b_im, temp_mout, 1280, 1000, 1000,
        1000, 1000, 1000, 0, 0, 0, 0);
    gemm_bf16_wmma<false><<<gemm_grid(1280, 1000, 1), 256, 0, stream>>>(
        att_tempo, W_it, b_it, temp_tout, 1280, 1000, 1000,
        1000, 1000, 1000, 0, 0, 0, 0);

    // ---- Stage 2: softmax-gating + concat into mm_inp [1280,2300] ----
    softmax_mul_concat_kernel<<<1280, 256, 0, stream>>>(embeddings, temp_mout, temp_tout, mm_inp);

    // ---- Stage 3: mm_out = mm_inp @ W_mm^T + b_mm ----
    gemm_bf16_wmma<false><<<gemm_grid(1280, 1024, 1), 256, 0, stream>>>(
        mm_inp, W_mm, b_mm, mm_out, 1280, 1024, 2300,
        2300, 2300, 1024, 0, 0, 0, 0);

    // ---- Stage 4: mean pooling + small init linear m0 = mean_feat @ W_li^T ----
    mean_pool_kernel<<<(64 * 2 * 200 + 255) / 256, 256, 0, stream>>>(att_tempo, att_motion, mean_feat);
    gemm_bf16_wmma<false><<<gemm_grid(128, 1024, 1), 256, 0, stream>>>(
        mean_feat, W_li, b_li, m0, 128, 1024, 200,
        200, 200, 1024, 0, 0, 0, 0);

    // ---- Stage 5: LSTMs (shared weights). bcomb = bih + bhh ----
    bias_comb_kernel<<<(4096 + 255) / 256, 256, 0, stream>>>(bih, bhh, bcomb, 4096);
    zero_kernel<<<(64 * 1024 + 255) / 256, 256, 0, stream>>>(hbuf, 64 * 1024);
    zero_kernel<<<(64 * 1024 + 255) / 256, 256, 0, stream>>>(cbuf, 64 * 1024);

    // warm-up LSTM over m0 (T=2) -> (h0, c0)
    for (int t = 0; t < 2; ++t) {
        gemm_bf16_wmma<false><<<gemm_grid(64, 4096, 1), 256, 0, stream>>>(
            m0 + (long long)t * 1024, Wih, bcomb, gates, 64, 4096, 1024,
            2048, 1024, 4096, 0, 0, 0, 0);
        gemm_bf16_wmma<false><<<gemm_grid(64, 4096, 1), 256, 0, stream>>>(
            hbuf, Whh, nullptr, gates, 64, 4096, 1024,
            1024, 1024, 4096, 0, 0, 0, 1);
        lstm_cell_kernel<<<(64 * 1024 + 255) / 256, 256, 0, stream>>>(gates, cbuf, hbuf, nullptr, 0);
    }
    // main LSTM over mm_out (T=20), act = tanh(h_t)
    for (int t = 0; t < 20; ++t) {
        gemm_bf16_wmma<false><<<gemm_grid(64, 4096, 1), 256, 0, stream>>>(
            mm_out + (long long)t * 1024, Wih, bcomb, gates, 64, 4096, 1024,
            20480, 1024, 4096, 0, 0, 0, 0);
        gemm_bf16_wmma<false><<<gemm_grid(64, 4096, 1), 256, 0, stream>>>(
            hbuf, Whh, nullptr, gates, 64, 4096, 1024,
            1024, 1024, 4096, 0, 0, 0, 1);
        lstm_cell_kernel<<<(64 * 1024 + 255) / 256, 256, 0, stream>>>(
            gates, cbuf, hbuf, act + (long long)t * 1024, 20480);
    }

    // ---- Stage 6: masked attention (motion) ----
    rowdot_kernel<<<1280, 256, 0, stream>>>(act, w_ms_e, eproj, 1024);
    rowdot_kernel<<<64 * 200, 256, 0, stream>>>(mfeat, w_ms_f, fproj, 1000);
    attn_weights_kernel<<<1280, 256, 0, stream>>>(eproj, fproj, mlens, b_ms, wmat);
    gemm_bf16_wmma<true><<<gemm_grid(20, 1000, 64), 256, 0, stream>>>(
        wmat, mfeat, nullptr, att_m2, 20, 1000, 200,
        200, 1000, 1000, 20 * 200, 200 * 1000, 20 * 1000, 0);

    // ---- Stage 7: masked attention (tempo) ----
    rowdot_kernel<<<1280, 256, 0, stream>>>(act, w_ts_e, eproj, 1024);
    rowdot_kernel<<<64 * 200, 256, 0, stream>>>(tfeat, w_ts_f, fproj, 1000);
    attn_weights_kernel<<<1280, 256, 0, stream>>>(eproj, fproj, tlens, b_ts, wmat);
    gemm_bf16_wmma<true><<<gemm_grid(20, 1000, 64), 256, 0, stream>>>(
        wmat, tfeat, nullptr, att_t2, 20, 1000, 200,
        200, 1000, 1000, 20 * 200, 200 * 1000, 20 * 1000, 0);

    // ---- Stage 8: final projection (dominant GEMM) + in-place log_softmax ----
    concat_final_kernel<<<(unsigned)((1280LL * 3024 + 255) / 256), 256, 0, stream>>>(
        act, att_t2, att_m2, Afin);
    gemm_bf16_wmma<false><<<gemm_grid(1280, 20000, 1), 256, 0, stream>>>(
        Afin, W_out, b_out, out, 1280, 20000, 3024,
        3024, 3024, 20000, 0, 0, 0, 0);
    log_softmax_kernel<<<1280, 256, 0, stream>>>(out, 20000);
}
